// GAT_30580167148118
// MI455X (gfx1250) — matrix operand order repaired and checked by reading, not compiled, not run on hardware
//
#include <hip/hip_runtime.h>
#include <hip/hip_bf16.h>
#include <math.h>

// ---------------- constants from the reference ----------------
#define NN   100000
#define GG   512
#define HID  48
#define INC  64
#define NH   4
#define CC   12          // HID / NH
#define OUTC 10

typedef __attribute__((ext_vector_type(16))) _Float16 v16h;
typedef __attribute__((ext_vector_type(8)))  float    v8f;

// Order-preserving float<->uint mapping so atomicMax(u32) == float max.
__device__ __forceinline__ unsigned f2ord(float f) {
    unsigned u = __float_as_uint(f);
    return (u & 0x80000000u) ? ~u : (u | 0x80000000u);
}
__device__ __forceinline__ float ord2f(unsigned o) {
    unsigned u = (o & 0x80000000u) ? (o & 0x7fffffffu) : ~o;
    return __uint_as_float(u);
}

// =====================================================================
// WMMA GEMM, compile-time shapes:  Y[Nrows,NCOLS] = act(X[Nrows,K]) @ W[K,NCOLS]
//  - one wave32 per 16-row strip; it computes ALL col tiles (A reuse)
//  - A: two aligned float4 loads per in-range 16-wide K segment (no guards)
//  - B: staged in LDS, zero-padded to KP x NCP, unguarded ds reads
//  - requires Nrows % 16 == 0 (holds: 100000, 512)
// =====================================================================
template<int K, int NCOLS, bool RELU>
__global__ __launch_bounds__(32)
void gemm_wmma_tiled(const float* __restrict__ X, const float* __restrict__ W,
                     float* __restrict__ Y)
{
    constexpr int KP  = ((K + 31) / 32) * 32;       // K padded to 32
    constexpr int NCP = ((NCOLS + 15) / 16) * 16;   // cols padded to 16
    constexpr int NT  = NCP / 16;                   // col tiles

    __shared__ float Ws[KP * NCP];

    const int lane = threadIdx.x;       // 0..31
    const int lc   = lane & 15;
    const int half = lane >> 4;
    const int bm   = blockIdx.x * 16;
    const int row  = bm + lc;

    // ---- stage W into LDS (coalesced float4; pad with zeros) ----
    if constexpr (NCOLS == NCP) {
        const float4* Wv = (const float4*)W;
        float4* Sv = (float4*)Ws;
        for (int t = lane; t * 4 < K * NCP; t += 32) Sv[t] = Wv[t];
        float4 z4 = {0.f, 0.f, 0.f, 0.f};
        float4* Zv = (float4*)(Ws + K * NCP);
        for (int t = lane; t * 4 < (KP - K) * NCP; t += 32) Zv[t] = z4;
    } else {
        for (int t = lane; t < KP * NCP; t += 32) {
            int k = t / NCP, c = t % NCP;
            Ws[t] = (k < K && c < NCOLS) ? W[k * NCOLS + c] : 0.f;
        }
    }
    __syncthreads();

    v8f acc[NT];
#pragma unroll
    for (int ct = 0; ct < NT; ++ct) acc[ct] = (v8f){};

    const float* xrow = X + (long)row * K;

#pragma unroll
    for (int kb = 0; kb < KP; kb += 32) {
        // ---- A fragment: lanes 0-15 row M, K kb+{half*8..+7, 16+half*8..+7} ----
        v16h a;
#pragma unroll
        for (int seg = 0; seg < 2; ++seg) {
            if constexpr (true) {}  // keep scope
            if ((kb + seg * 16) < K) {   // compile-time (kb, seg, K constants)
                const float* ap = xrow + kb + seg * 16 + half * 8;
                float4 lo = *(const float4*)(ap);
                float4 hi = *(const float4*)(ap + 4);
                float f0 = lo.x, f1 = lo.y, f2 = lo.z, f3 = lo.w;
                float f4 = hi.x, f5 = hi.y, f6 = hi.z, f7 = hi.w;
                if (RELU) {
                    f0 = fmaxf(f0, 0.f); f1 = fmaxf(f1, 0.f); f2 = fmaxf(f2, 0.f); f3 = fmaxf(f3, 0.f);
                    f4 = fmaxf(f4, 0.f); f5 = fmaxf(f5, 0.f); f6 = fmaxf(f6, 0.f); f7 = fmaxf(f7, 0.f);
                }
                a[seg * 8 + 0] = (_Float16)f0; a[seg * 8 + 1] = (_Float16)f1;
                a[seg * 8 + 2] = (_Float16)f2; a[seg * 8 + 3] = (_Float16)f3;
                a[seg * 8 + 4] = (_Float16)f4; a[seg * 8 + 5] = (_Float16)f5;
                a[seg * 8 + 6] = (_Float16)f6; a[seg * 8 + 7] = (_Float16)f7;
            } else {
#pragma unroll
                for (int i = 0; i < 8; ++i) a[seg * 8 + i] = (_Float16)0.f;
            }
        }
        // ---- B fragments from LDS; 3 WMMAs reuse the A fragment ----
#pragma unroll
        for (int ct = 0; ct < NT; ++ct) {
            const int col = ct * 16 + lc;
            v16h b;
#pragma unroll
            for (int e = 0; e < 16; ++e) {
                b[e] = (_Float16)Ws[(kb + (e >> 3) * 16 + half * 8 + (e & 7)) * NCP + col];
            }
            acc[ct] = __builtin_amdgcn_wmma_f32_16x16x32_f16(
                false, a, false, b, (short)0, acc[ct], false, false);
        }
    }

    // ---- store D: VGPR v -> row bm + v + half*8, col = ct*16 + lc ----
#pragma unroll
    for (int ct = 0; ct < NT; ++ct) {
        const int col = ct * 16 + lc;
        if (NCOLS % 16 == 0 || col < NCOLS) {
#pragma unroll
            for (int v = 0; v < 8; ++v) {
                int r = bm + v + half * 8;
                Y[(long)r * NCOLS + col] = acc[ct][v];
            }
        }
    }
}

// ---------------- per-(node,head) attention coefficients ----------------
__global__ void attn_coef_kernel(const float* __restrict__ xh,
                                 const float* __restrict__ a_src,
                                 const float* __restrict__ a_dst,
                                 float* __restrict__ als, float* __restrict__ ald)
{
    long t = (long)blockIdx.x * blockDim.x + threadIdx.x;
    if (t >= (long)NN * NH) return;
    int n = (int)(t >> 2), h = (int)(t & 3);
    const float* xr = xh + (long)n * HID + h * CC;
    const float* av = a_src + h * CC;
    const float* bv = a_dst + h * CC;
    float ss = 0.f, sd = 0.f;
#pragma unroll
    for (int c = 0; c < CC; ++c) { float v = xr[c]; ss += v * av[c]; sd += v * bv[c]; }
    als[t] = ss;
    ald[t] = sd;
}

__device__ __forceinline__ void edge_ends(const int* ei, int E, int e, int& s, int& d) {
    if (e < E) { s = ei[e]; d = ei[E + e]; } else { s = e - E; d = s; }  // self loops appended
}

// ---------------- pass 1: segment max of leaky-relu logits ----------------
__global__ void edge_max_kernel(const int* __restrict__ ei, int E, int EE,
                                const float* __restrict__ als, const float* __restrict__ ald,
                                unsigned* __restrict__ mmax)
{
    long t = (long)blockIdx.x * blockDim.x + threadIdx.x;
    if (t >= (long)EE * NH) return;
    int e = (int)(t >> 2), h = (int)(t & 3);
    int s, d; edge_ends(ei, E, e, s, d);
    float l = als[(long)s * NH + h] + ald[(long)d * NH + h];
    l = (l > 0.f) ? l : 0.2f * l;
    atomicMax(&mmax[(long)d * NH + h], f2ord(l));
}

// ---------------- pass 2: denom = segment sum of exp(logit - max) ----------------
__global__ void edge_sum_kernel(const int* __restrict__ ei, int E, int EE,
                                const float* __restrict__ als, const float* __restrict__ ald,
                                const unsigned* __restrict__ mmax,
                                float* __restrict__ denom,
                                const float* __restrict__ xh)
{
    long t = (long)blockIdx.x * blockDim.x + threadIdx.x;
    if (t >= (long)EE * NH) return;
    int e = (int)(t >> 2), h = (int)(t & 3);
    int s, d; edge_ends(ei, E, e, s, d);
    float l = als[(long)s * NH + h] + ald[(long)d * NH + h];
    l = (l > 0.f) ? l : 0.2f * l;
    float ex = __expf(l - ord2f(mmax[(long)d * NH + h]));
    atomicAdd(&denom[(long)d * NH + h], ex);
    // warm L2 with the source row that pass 3 will gather (global_prefetch_b8)
    __builtin_prefetch(xh + (long)s * HID + h * CC, 0, 1);
}

// ---------------- pass 3: weighted scatter-add of messages ----------------
__global__ void edge_agg_kernel(const int* __restrict__ ei, int E, int EE,
                                const float* __restrict__ als, const float* __restrict__ ald,
                                const unsigned* __restrict__ mmax,
                                const float* __restrict__ denom,
                                const float* __restrict__ xh,
                                float* __restrict__ hout)
{
    long t = (long)blockIdx.x * blockDim.x + threadIdx.x;
    if (t >= (long)EE * NH) return;
    int e = (int)(t >> 2), h = (int)(t & 3);
    int s, d; edge_ends(ei, E, e, s, d);
    float l = als[(long)s * NH + h] + ald[(long)d * NH + h];
    l = (l > 0.f) ? l : 0.2f * l;
    float ex = __expf(l - ord2f(mmax[(long)d * NH + h]));
    float alpha = ex / (denom[(long)d * NH + h] + 1e-16f);
    const float* xs = xh + (long)s * HID + h * CC;
    float* od = hout + (long)d * HID + h * CC;
#pragma unroll
    for (int c = 0; c < CC; ++c) atomicAdd(&od[c], xs[c] * alpha);
}

// ---------------- global mean pool (reads relu(h)) ----------------
__global__ void pool_kernel(const float* __restrict__ h, const int* __restrict__ batch,
                            float* __restrict__ sums, float* __restrict__ cnt)
{
    long t = (long)blockIdx.x * blockDim.x + threadIdx.x;
    if (t >= (long)NN * NH) return;
    int n = (int)(t >> 2), q = (int)(t & 3);
    int b = batch[n];
    const float* hr = h + (long)n * HID + q * CC;
    float* sr = sums + (long)b * HID + q * CC;
#pragma unroll
    for (int c = 0; c < CC; ++c) atomicAdd(&sr[c], fmaxf(hr[c], 0.f));
    if (q == 0) atomicAdd(&cnt[b], 1.0f);
}

__global__ void pool_div_kernel(const float* __restrict__ sums, const float* __restrict__ cnt,
                                float* __restrict__ g)
{
    int t = blockIdx.x * blockDim.x + threadIdx.x;
    if (t >= GG * HID) return;
    g[t] = sums[t] / fmaxf(cnt[t / HID], 1.0f);
}

// ---------------- batchnorm stats over the G dim (mb1 cancels under mean-sub) ----
__global__ void bn_stats_kernel(const float* __restrict__ z, float* __restrict__ mu,
                                float* __restrict__ var)
{
    __shared__ float s1[256], s2[256];
    int j = blockIdx.x, tid = threadIdx.x;
    float a = 0.f, b = 0.f;
    for (int g = tid; g < GG; g += 256) { float v = z[(long)g * HID + j]; a += v; b += v * v; }
    s1[tid] = a; s2[tid] = b;
    __syncthreads();
    for (int st = 128; st > 0; st >>= 1) {
        if (tid < st) { s1[tid] += s1[tid + st]; s2[tid] += s2[tid + st]; }
        __syncthreads();
    }
    if (tid == 0) {
        float m = s1[0] / (float)GG;
        mu[j] = m;
        var[j] = s2[0] / (float)GG - m * m;
    }
}

__global__ void bn_apply_kernel(float* __restrict__ z, const float* __restrict__ mu,
                                const float* __restrict__ var, const float* __restrict__ gamma,
                                const float* __restrict__ beta)
{
    int t = blockIdx.x * blockDim.x + threadIdx.x;
    if (t >= GG * HID) return;
    int j = t % HID;
    float v = (z[t] - mu[j]) * rsqrtf(var[j] + 1e-5f) * gamma[j] + beta[j];
    z[t] = fmaxf(v, 0.f);   // ReLU
}

// ---------------- final log_softmax / softmax / logits ----------------
__global__ void softmax_kernel(const float* __restrict__ lastraw, const float* __restrict__ mb2,
                               float* __restrict__ out)
{
    int g = blockIdx.x * blockDim.x + threadIdx.x;
    if (g >= GG) return;
    float v[OUTC];
    float mx = -1e30f;
#pragma unroll
    for (int o = 0; o < OUTC; ++o) { v[o] = lastraw[(long)g * OUTC + o] + mb2[o]; mx = fmaxf(mx, v[o]); }
    float s = 0.f;
#pragma unroll
    for (int o = 0; o < OUTC; ++o) s += __expf(v[o] - mx);
    float ls = __logf(s);
#pragma unroll
    for (int o = 0; o < OUTC; ++o) {
        float lp = v[o] - mx - ls;
        out[(long)g * OUTC + o]                   = lp;            // logp
        out[(long)GG * OUTC + (long)g * OUTC + o] = __expf(lp);    // soft
        out[2L * GG * OUTC + (long)g * OUTC + o]  = v[o];          // last
    }
}

// =======================================================================
extern "C" void kernel_launch(void* const* d_in, const int* in_sizes, int n_in,
                              void* d_out, int out_size, void* d_ws, size_t ws_size,
                              hipStream_t stream)
{
    const float* x     = (const float*)d_in[0];
    const int*   ei    = (const int*)  d_in[1];
    const int*   batch = (const int*)  d_in[3];
    const float* Wg[3]  = { (const float*)d_in[4],  (const float*)d_in[7],  (const float*)d_in[10] };
    const float* aS[3]  = { (const float*)d_in[5],  (const float*)d_in[8],  (const float*)d_in[11] };
    const float* aD[3]  = { (const float*)d_in[6],  (const float*)d_in[9],  (const float*)d_in[12] };
    const float* mW1   = (const float*)d_in[13];
    const float* gamma = (const float*)d_in[15];
    const float* beta  = (const float*)d_in[16];
    const float* mW2   = (const float*)d_in[17];
    const float* mb2   = (const float*)d_in[18];

    const int E  = in_sizes[1] / 2;       // edges before self loops
    const int EE = E + NN;                // with self loops

    // -------- workspace layout (floats) --------
    float*    xh    = (float*)d_ws;                 // N*HID
    float*    hA    = xh   + (long)NN * HID;        // N*HID
    float*    hB    = hA   + (long)NN * HID;        // N*HID
    float*    als   = hB   + (long)NN * HID;        // N*NH
    float*    ald   = als  + (long)NN * NH;         // N*NH
    unsigned* mmax  = (unsigned*)(ald + (long)NN * NH);   // N*NH
    float*    denom = (float*)mmax + (long)NN * NH;       // N*NH
    float*    sums  = denom + (long)NN * NH;        // G*HID
    float*    cnt   = sums  + (long)GG * HID;       // G
    float*    gbuf  = cnt   + GG;                   // G*HID
    float*    zbuf  = gbuf  + (long)GG * HID;       // G*HID
    float*    mu    = zbuf  + (long)GG * HID;       // HID
    float*    var   = mu    + HID;                  // HID
    float*    lastr = var   + HID;                  // G*OUTC
    float*    outp  = (float*)d_out;

    const int TPB = 256;
    const long eth = (long)EE * NH;
    const int  eblocks = (int)((eth + TPB - 1) / TPB);
    const int  nblocks = (int)(((long)NN * NH + TPB - 1) / TPB);

    // -------- three GAT layers --------
    for (int L = 0; L < 3; ++L) {
        const float* hin  = (L == 0) ? x  : ((L == 1) ? hA : hB);
        float*       hout = (L == 1) ? hB : hA;     // L0->hA, L1->hB, L2->hA

        if (L == 0)
            gemm_wmma_tiled<INC, HID, false><<<NN / 16, 32, 0, stream>>>(hin, Wg[L], xh);
        else
            gemm_wmma_tiled<HID, HID, true><<<NN / 16, 32, 0, stream>>>(hin, Wg[L], xh);

        attn_coef_kernel<<<nblocks, TPB, 0, stream>>>(xh, aS[L], aD[L], als, ald);

        hipMemsetAsync(mmax,  0, (size_t)NN * NH * sizeof(unsigned), stream);
        hipMemsetAsync(denom, 0, (size_t)NN * NH * sizeof(float),    stream);
        hipMemsetAsync(hout,  0, (size_t)NN * HID * sizeof(float),   stream);

        edge_max_kernel<<<eblocks, TPB, 0, stream>>>(ei, E, EE, als, ald, mmax);
        edge_sum_kernel<<<eblocks, TPB, 0, stream>>>(ei, E, EE, als, ald, mmax, denom, xh);
        edge_agg_kernel<<<eblocks, TPB, 0, stream>>>(ei, E, EE, als, ald, mmax, denom, xh, hout);
    }

    // -------- global mean pool (applies final ReLU on read) --------
    hipMemsetAsync(sums, 0, (size_t)GG * HID * sizeof(float), stream);
    hipMemsetAsync(cnt,  0, (size_t)GG * sizeof(float),       stream);
    pool_kernel<<<nblocks, TPB, 0, stream>>>(hA, batch, sums, cnt);
    pool_div_kernel<<<(GG * HID + TPB - 1) / TPB, TPB, 0, stream>>>(sums, cnt, gbuf);

    // -------- MLP head --------
    gemm_wmma_tiled<HID, HID, false><<<GG / 16, 32, 0, stream>>>(gbuf, mW1, zbuf);
    bn_stats_kernel<<<HID, 256, 0, stream>>>(zbuf, mu, var);
    bn_apply_kernel<<<(GG * HID + TPB - 1) / TPB, TPB, 0, stream>>>(zbuf, mu, var, gamma, beta);
    gemm_wmma_tiled<HID, OUTC, false><<<GG / 16, 32, 0, stream>>>(zbuf, mW2, lastr);
    softmax_kernel<<<(GG + TPB - 1) / TPB, TPB, 0, stream>>>(lastr, mb2, outp);
}
